// LSHAttention_66365834658174
// MI455X (gfx1250) — matrix operand order repaired and checked by reading, not compiled, not run on hardware
//
#include <hip/hip_runtime.h>
#include <cmath>

#define BATCH 32
#define SEQ   2048
#define DK    64
#define NR    4
#define NB    32
#define BL    64    // SEQ / NB
#define KL    128   // 2 * BL

typedef __attribute__((ext_vector_type(16))) _Float16 v16h;
typedef __attribute__((ext_vector_type(8)))  _Float16 v8h;
typedef __attribute__((ext_vector_type(4)))  _Float16 v4h;
typedef __attribute__((ext_vector_type(8)))  float    v8f;

union HVec { v16h v; v8h h[2]; _Float16 e[16]; };

// ---------------- K1: projection + argmax bucket hash ----------------
// packed[b*SEQ+s] holds the 4 round buckets, one byte each.
__global__ __launch_bounds__(256) void lsh_hash_kernel(const float* __restrict__ query,
                                                       const float* __restrict__ randm,
                                                       unsigned* __restrict__ packed) {
  __shared__ float s_rand[DK * NR * (NB / 2)];   // 4096 floats = 16 KB
  const int b = blockIdx.y;
  const int t = threadIdx.x;
  for (int i = t; i < DK * NR * (NB / 2); i += 256)
    s_rand[i] = randm[(size_t)b * DK * NR * (NB / 2) + i];
  __syncthreads();

  const int s = blockIdx.x * 256 + t;
  const float* qp = query + ((size_t)b * SEQ + s) * DK;
  float q[DK];
  #pragma unroll
  for (int d4 = 0; d4 < DK / 4; ++d4) {
    float4 f = ((const float4*)qp)[d4];
    q[d4 * 4 + 0] = f.x; q[d4 * 4 + 1] = f.y; q[d4 * 4 + 2] = f.z; q[d4 * 4 + 3] = f.w;
  }

  unsigned pk = 0;
  for (int r = 0; r < NR; ++r) {
    float pr[NB / 2];
    for (int k = 0; k < NB / 2; ++k) {
      float acc = 0.f;
      #pragma unroll 8
      for (int d = 0; d < DK; ++d) acc += q[d] * s_rand[(d * NR + r) * (NB / 2) + k];
      pr[k] = acc;
    }
    float best = -1e30f; int bi = 0;
    for (int i = 0; i < NB; ++i) {          // first-max wins, like jnp.argmax
      float v = (i < NB / 2) ? pr[i] : -pr[i - NB / 2];
      if (v > best) { best = v; bi = i; }
    }
    pk |= ((unsigned)bi) << (8 * r);
  }
  packed[b * SEQ + s] = pk;
}

// ---------------- K2: stable counting sort per (b, r) ----------------
// One wave; lane L owns bucket L and scans positions in order (stable).
__global__ __launch_bounds__(32) void lsh_sort_kernel(const unsigned* __restrict__ packed,
                                                      int* __restrict__ sort_idx) {
  __shared__ unsigned s_pk[SEQ];   // 8 KB (pre-shifted bucket byte)
  __shared__ int s_cnt[NB];
  const int r = blockIdx.x, b = blockIdx.y;
  const int lane = threadIdx.x;
  for (int i = lane; i < SEQ; i += 32)
    s_pk[i] = (packed[b * SEQ + i] >> (8 * r)) & 0xFFu;
  __syncthreads();
  int c = 0;
  for (int i = 0; i < SEQ; ++i) c += (s_pk[i] == (unsigned)lane);
  s_cnt[lane] = c;
  __syncthreads();
  int base = 0;
  for (int j = 0; j < lane; ++j) base += s_cnt[j];
  int* out = sort_idx + ((size_t)b * NR + r) * SEQ;
  int p = base;
  for (int i = 0; i < SEQ; ++i)
    if (s_pk[i] == (unsigned)lane) out[p++] = i;
}

// ---------------- K3: blocked LSH attention (WMMA) ----------------
__global__ __launch_bounds__(256) void lsh_attn_kernel(const float* __restrict__ query,
                                                       const float* __restrict__ value,
                                                       const int* __restrict__ sort_idx,
                                                       const unsigned* __restrict__ packed,
                                                       float* __restrict__ lse_out,
                                                       float* __restrict__ att_out) {
  __shared__ int      s_qi[BL];
  __shared__ int      s_ki[KL];
  __shared__ unsigned s_kb[KL];
  __shared__ float    s_cnt[KL];
  __shared__ __align__(16) _Float16 s_q[BL * DK];    //  8 KB  A for QK, row-major [m][d]
  __shared__ __align__(16) _Float16 s_k[KL * DK];    // 16 KB  normalized keys [n][d]; reused as SM
  __shared__ __align__(16) _Float16 s_vt[DK * KL];   // 16 KB  V transposed [d][j]
  __shared__ float s_qk[BL * KL];                    // 32 KB  f32 logits

  const int nb = blockIdx.x, r = blockIdx.y, b = blockIdx.z;
  const int t = threadIdx.x;
  const int w = t >> 5, lane = t & 31;
  const int lm = lane & 15, lh = lane >> 4;
  const int* sidx = sort_idx + ((size_t)b * NR + r) * SEQ;

  // --- phase A: indices ---
  if (t < BL) {
    s_qi[t] = sidx[nb * BL + t];
  } else if (t < BL + KL) {
    int j = t - BL;
    int pb = (j < BL) ? ((nb + NB - 1) & (NB - 1)) : nb;   // prev bucket first
    int oi = sidx[pb * BL + (j & (BL - 1))];
    s_ki[j] = oi;
    s_kb[j] = packed[b * SEQ + oi];
  }
  __syncthreads();

  // --- phase B: per-column duplicate counts + gathers ---
  if (t < KL) {
    unsigned kp = s_kb[t];
    int c = 0;
    #pragma unroll
    for (int rr = 0; rr < NR; ++rr) {
      int kb = (kp >> (8 * rr)) & 0xFF;
      c += (kb == nb) || (kb == ((nb + NB - 1) & (NB - 1)));
    }
    s_cnt[t] = (float)c;   // >= 1 always (own round contributes)
  }
  // gather Q rows (f16, row-major), vectorized 4-wide
  for (int e = t; e < BL * DK / 4; e += 256) {
    int row = e >> 4, d0 = (e & 15) * 4;
    float4 f = *(const float4*)(query + ((size_t)b * SEQ + s_qi[row]) * DK + d0);
    v4h hv = { (_Float16)f.x, (_Float16)f.y, (_Float16)f.z, (_Float16)f.w };
    *(v4h*)(s_q + row * DK + d0) = hv;
  }
  // gather V rows transposed into [d][j]
  for (int e = t; e < KL * DK / 4; e += 256) {
    int row = e >> 4, d0 = (e & 15) * 4;
    float4 f = *(const float4*)(value + ((size_t)b * SEQ + s_ki[row]) * DK + d0);
    s_vt[(d0 + 0) * KL + row] = (_Float16)f.x;
    s_vt[(d0 + 1) * KL + row] = (_Float16)f.y;
    s_vt[(d0 + 2) * KL + row] = (_Float16)f.z;
    s_vt[(d0 + 3) * KL + row] = (_Float16)f.w;
  }
  {
    // keys = gathered queries, L2-normalized, row-major [n][d]
    int row = t >> 1, half = t & 1;
    const float* kp = query + ((size_t)b * SEQ + s_ki[row]) * DK + half * 32;
    float v32[32]; float ss = 0.f;
    #pragma unroll
    for (int i4 = 0; i4 < 8; ++i4) {
      float4 f = ((const float4*)kp)[i4];
      v32[i4 * 4 + 0] = f.x; v32[i4 * 4 + 1] = f.y;
      v32[i4 * 4 + 2] = f.z; v32[i4 * 4 + 3] = f.w;
      ss += f.x * f.x + f.y * f.y + f.z * f.z + f.w * f.w;
    }
    ss += __shfl_xor(ss, 1);
    float sc = rsqrtf(fmaxf(ss, 1e-12f));
    _Float16* dst = s_k + row * DK + half * 32;
    #pragma unroll
    for (int i8 = 0; i8 < 4; ++i8) {
      v8h hv;
      #pragma unroll
      for (int i = 0; i < 8; ++i) hv[i] = (_Float16)(v32[i8 * 8 + i] * sc);
      *(v8h*)(dst + i8 * 8) = hv;
    }
  }
  __syncthreads();

  // --- phase C: QK = Q * K^T / 8 via v_wmma_f32_16x16x32_f16 ---
  {
    const int nT = w;                               // 8 waves -> 8 N-tiles of 16
    for (int mT = 0; mT < 4; ++mT) {
      v8f acc = {};
      #pragma unroll
      for (int kc = 0; kc < 2; ++kc) {
        HVec A, Bv;
        const _Float16* ap = s_q + (mT * 16 + lm) * DK + kc * 32 + lh * 8;
        A.h[0] = *(const v8h*)ap;                   // K = kc*32 + lh*8 .. +7
        A.h[1] = *(const v8h*)(ap + 16);            // K = kc*32 + 16 + lh*8 ..
        // B[k][n] = k_norm[n][k] -> contiguous in row-major keys
        const _Float16* bp = s_k + (nT * 16 + lm) * DK + kc * 32 + lh * 8;
        Bv.h[0] = *(const v8h*)bp;
        Bv.h[1] = *(const v8h*)(bp + 16);
        acc = __builtin_amdgcn_wmma_f32_16x16x32_f16(false, A.v, false, Bv.v,
                                                     (short)0, acc, false, false);
      }
      const int n_abs = nT * 16 + lm;
      const int kiv = s_ki[n_abs];
      #pragma unroll
      for (int rr = 0; rr < 8; ++rr) {
        const int m_abs = mT * 16 + rr + lh * 8;
        const int qiv = s_qi[m_abs];
        float v = acc[rr] * 0.125f;                 // / sqrt(64)
        v = (qiv < kiv) ? -1e9f : ((qiv == kiv) ? -1e5f : v);
        s_qk[m_abs * KL + n_abs] = v;
      }
    }
  }
  __syncthreads();

  // --- phase D: softmax over 128 keys, / counts, LSE scatter ---
  _Float16* s_sm = s_k;   // keys dead after phase C; reuse 16 KB as SM [m][j]
  {
    const int row = t >> 2, seg = t & 3;
    const float* qrow = s_qk + row * KL + seg * 32;
    float mx = -1e30f;
    #pragma unroll
    for (int i = 0; i < 32; ++i) mx = fmaxf(mx, qrow[i]);
    mx = fmaxf(mx, __shfl_xor(mx, 1));
    mx = fmaxf(mx, __shfl_xor(mx, 2));
    float se = 0.f;
    #pragma unroll
    for (int i = 0; i < 32; ++i) se += __expf(qrow[i] - mx);
    se += __shfl_xor(se, 1);
    se += __shfl_xor(se, 2);
    const float inv = 1.f / se;
    #pragma unroll
    for (int i = 0; i < 32; ++i) {
      int j = seg * 32 + i;
      s_sm[row * KL + j] = (_Float16)(__expf(qrow[i] - mx) * inv / s_cnt[j]);
    }
    if (seg == 0)
      lse_out[((size_t)b * NR + r) * SEQ + s_qi[row]] = mx + __logf(se);
  }
  __syncthreads();

  // --- phase E: O = SM * V via WMMA; scatter rows to original positions ---
  {
    #pragma unroll
    for (int q2 = 0; q2 < 2; ++q2) {
      const int ti = w * 2 + q2;                    // 16 tiles over 8 waves
      const int mT = ti >> 2, nT = ti & 3;
      v8f acc = {};
      #pragma unroll
      for (int jc = 0; jc < 4; ++jc) {
        HVec A, Bv;
        const _Float16* ap = s_sm + (mT * 16 + lm) * KL + jc * 32 + lh * 8;
        A.h[0] = *(const v8h*)ap;
        A.h[1] = *(const v8h*)(ap + 16);
        // B[k][n] = v[jc*32+k][n] = vt[n][jc*32+k] -> contiguous in vt rows
        const _Float16* bp = s_vt + (nT * 16 + lm) * KL + jc * 32 + lh * 8;
        Bv.h[0] = *(const v8h*)bp;
        Bv.h[1] = *(const v8h*)(bp + 16);
        acc = __builtin_amdgcn_wmma_f32_16x16x32_f16(false, A.v, false, Bv.v,
                                                     (short)0, acc, false, false);
      }
      const int d = nT * 16 + lm;
      #pragma unroll
      for (int rr = 0; rr < 8; ++rr) {
        const int m_abs = mT * 16 + rr + lh * 8;
        att_out[(((size_t)b * NR + r) * SEQ + s_qi[m_abs]) * DK + d] = acc[rr];
      }
    }
  }
}

// ---------------- K4: combine rounds with softmax(LSE) weights ----------------
__global__ __launch_bounds__(256) void lsh_combine_kernel(const float* __restrict__ att,
                                                          const float* __restrict__ lse,
                                                          float* __restrict__ out) {
  const int b = blockIdx.y;
  const int t = threadIdx.x;
  const int d = t & 63;
  const int s = blockIdx.x * 4 + (t >> 6);
  float l[NR];
  #pragma unroll
  for (int r = 0; r < NR; ++r) l[r] = lse[((size_t)b * NR + r) * SEQ + s];
  float mx = l[0];
  #pragma unroll
  for (int r = 1; r < NR; ++r) mx = fmaxf(mx, l[r]);
  float wsum = 0.f, wr[NR];
  #pragma unroll
  for (int r = 0; r < NR; ++r) { wr[r] = __expf(l[r] - mx); wsum += wr[r]; }
  float o = 0.f;
  #pragma unroll
  for (int r = 0; r < NR; ++r)
    o += att[(((size_t)b * NR + r) * SEQ + s) * DK + d] * wr[r];
  out[((size_t)b * SEQ + s) * DK + d] = o / wsum;
}

extern "C" void kernel_launch(void* const* d_in, const int* in_sizes, int n_in,
                              void* d_out, int out_size, void* d_ws, size_t ws_size,
                              hipStream_t stream) {
  const float* query = (const float*)d_in[0];
  const float* value = (const float*)d_in[1];
  const float* randm = (const float*)d_in[2];
  float* out = (float*)d_out;

  // workspace layout (~66.3 MB total)
  char* ws = (char*)d_ws;
  int* sort_idx = (int*)ws;                                   // 1 MB
  size_t off = (size_t)BATCH * NR * SEQ * sizeof(int);
  unsigned* packed = (unsigned*)(ws + off);                   // 256 KB
  off += (size_t)BATCH * SEQ * sizeof(unsigned);
  float* lse = (float*)(ws + off);                            // 1 MB
  off += (size_t)BATCH * NR * SEQ * sizeof(float);
  float* att = (float*)(ws + off);                            // 64 MB

  lsh_hash_kernel   <<<dim3(SEQ / 256, BATCH),     256, 0, stream>>>(query, randm, packed);
  lsh_sort_kernel   <<<dim3(NR, BATCH),             32, 0, stream>>>(packed, sort_idx);
  lsh_attn_kernel   <<<dim3(NB, NR, BATCH),        256, 0, stream>>>(query, value, sort_idx,
                                                                     packed, lse, att);
  lsh_combine_kernel<<<dim3(SEQ / 4, BATCH),       256, 0, stream>>>(att, lse, out);
}
